// H0_48249662603534
// MI455X (gfx1250) — matrix operand order repaired and checked by reading, not compiled, not run on hardware
//
#include <hip/hip_runtime.h>
#include <hip/hip_bf16.h>

// ---------------------------------------------------------------------------
// Problem constants (from reference)
// ---------------------------------------------------------------------------
#define T_DIM 4096
#define B_DIM 8
#define F_DIM 3
#define L_DIM 4
#define S_DIM 8
#define D_DIM 32       // L*S
#define DI_DIM 64      // 2*D
#define DST_DIM 16     // mamba d_state
#define K_DIM 4
#define TB_DIM (T_DIM * B_DIM)      // 32768 tokens
#define LOG2PI 1.8378770664093453f

// ---------------------------------------------------------------------------
// WMMA vector types (CDNA5 wave32)
// ---------------------------------------------------------------------------
typedef __attribute__((ext_vector_type(16))) _Float16 v16h;
typedef __attribute__((ext_vector_type(8)))  _Float16 v8h;
typedef __attribute__((ext_vector_type(8)))  float    v8f;

// ---------------------------------------------------------------------------
// Workspace layout (bytes). Everything 256B-aligned; ~48 MB total.
// Weight fragments are stored pre-swizzled in WMMA per-lane register order:
//   frag[tile][lane][j] = W[k(j, lane)][col(tile, lane)]
// so a B operand is two contiguous b128 loads per lane (no LDS staging).
// ---------------------------------------------------------------------------
#define WS_H32    ((size_t)0)                    // [TB][32] f32 : h (HMM posteriors)
#define WS_H16    ((size_t)4194304)              // [TB][32] f16
#define WS_XIN    ((size_t)6291456)              // [TB][64] f32 : x branch of in-proj
#define WS_ZB     ((size_t)14680064)             // [TB][64] f32 : silu(z)
#define WS_XSS    ((size_t)23068672)             // [TB][64] f32 : post conv+silu
#define WS_XSS16  ((size_t)31457280)             // [TB][64] f16
#define WS_XDB    ((size_t)35651584)             // [TB][48] f32 : xss @ W_x (pad 34->48)
#define WS_YM16   ((size_t)41943040)             // [TB][64] f16 : gated SSM output
#define WS_MOUT   ((size_t)46137344)             // [TB][32] f32 : out-proj result
#define WS_WIN16  ((size_t)50331648)             // frag: 8 tiles  x32x16 = 8KB
#define WS_WX16   ((size_t)50339840)             // frag: 3x2 tiles x32x16 = 6KB
#define WS_WOUT16 ((size_t)50345984)             // frag: 2x2 tiles x32x16 = 4KB

// ---------------------------------------------------------------------------
// Helpers (fast paths: v_rcp_f32 / v_exp_f32 / v_log_f32, no IEEE divides)
// ---------------------------------------------------------------------------
__device__ __forceinline__ float sigmoidf_(float x) {
    return __builtin_amdgcn_rcpf(1.0f + __expf(-x));
}
__device__ __forceinline__ float siluf_(float x) { return x * sigmoidf_(x); }
__device__ __forceinline__ float softplusf_(float x) {
    return x > 20.0f ? x : __logf(1.0f + __expf(x));
}

// 16x32 A tile in this file's contiguous k order (same as B frags): half j of lane-group g holds
// K = g*16 + j  (two contiguous 16B packets per lane).
__device__ __forceinline__ v16h load_a_tile(const _Float16* __restrict__ A,
                                            int row, int lda, int k0, int g) {
    const _Float16* p = A + (size_t)row * lda + k0 + g * 16;
    v8h lo = *(const v8h*)(p);
    v8h hi = *(const v8h*)(p + 8);
    v16h a;
#pragma unroll
    for (int i = 0; i < 8; ++i) { a[i] = lo[i]; a[i + 8] = hi[i]; }
    return a;
}

// Pre-swizzled B fragment: 16 contiguous halves per lane = two b128 loads.
__device__ __forceinline__ v16h load_b_frag(const _Float16* __restrict__ frag,
                                            int tile, int lane) {
    const _Float16* p = frag + ((size_t)tile * 32 + lane) * 16;
    v8h lo = *(const v8h*)(p);
    v8h hi = *(const v8h*)(p + 8);
    v16h b;
#pragma unroll
    for (int i = 0; i < 8; ++i) { b[i] = lo[i]; b[i + 8] = hi[i]; }
    return b;
}

// ---------------------------------------------------------------------------
// K0: convert weights to f16 in WMMA B-fragment order.
// Contiguous k order chosen here (matches A loads): lane group g holds K = k0 + g*16 + j.
// ---------------------------------------------------------------------------
__global__ void prep_weights_kernel(const float* __restrict__ W_in,
                                    const float* __restrict__ W_x,
                                    const float* __restrict__ W_out,
                                    _Float16* __restrict__ win_frag,
                                    _Float16* __restrict__ wx_frag,
                                    _Float16* __restrict__ wout_frag) {
    int tid = threadIdx.x;
    // W_in [32][128] -> 8 tiles (ntile), K=32 (single wmma step)
    for (int i = tid; i < 8 * 32 * 16; i += 256) {
        int j = i & 15, lane = (i >> 4) & 31, ntile = i >> 9;
        int g = lane >> 4, n = lane & 15;
        int k = g * 16 + j;
        win_frag[i] = (_Float16)W_in[k * 128 + ntile * 16 + n];
    }
    // W_x [64][34] padded to 48 cols -> tiles (ntile 0..2, kc 0..1)
    for (int i = tid; i < 3 * 2 * 32 * 16; i += 256) {
        int j = i & 15, lane = (i >> 4) & 31, kc = (i >> 9) & 1, ntile = i >> 10;
        int g = lane >> 4, n = lane & 15;
        int k = kc * 32 + g * 16 + j;
        int col = ntile * 16 + n;
        wx_frag[i] = (col < 34) ? (_Float16)W_x[k * 34 + col] : (_Float16)0.0f;
    }
    // W_out [64][32] -> tiles (ntile 0..1, kc 0..1)
    for (int i = tid; i < 2 * 2 * 32 * 16; i += 256) {
        int j = i & 15, lane = (i >> 4) & 31, kc = (i >> 9) & 1, ntile = i >> 10;
        int g = lane >> 4, n = lane & 15;
        int k = kc * 32 + g * 16 + j;
        wout_frag[i] = (_Float16)W_out[k * 32 + ntile * 16 + n];
    }
}

// ---------------------------------------------------------------------------
// K1: fused EMA standardization + HMM log-forward. One wave per batch.
// lane = (l, s): l = lane>>3 layer, s = lane&7 state. Lanes s<3 also own the
// EMA channel for feature f=s of layer l. Cross-lane traffic via wave32 shfl.
// ---------------------------------------------------------------------------
__global__ void ema_hmm_scan_kernel(const float* __restrict__ x,
                                    const float* __restrict__ ema_alphas,
                                    const float* __restrict__ hmm_logA,
                                    const float* __restrict__ hmm_log_pi,
                                    const float* __restrict__ hmm_means,
                                    const float* __restrict__ hmm_log_vars,
                                    float* __restrict__ s_out,
                                    float* __restrict__ h32,
                                    _Float16* __restrict__ h16) {
    __shared__ __align__(16) float ldsT[L_DIM][S_DIM][S_DIM];
    const int lane = threadIdx.x;
    const int l = lane >> 3;
    const int s = lane & 7;
    const int b = blockIdx.x;

    // row-normalize transition logits: lane (l,s) handles row (l, i=s)
    {
        float row[S_DIM];
        float mx = -1e30f;
#pragma unroll
        for (int j = 0; j < S_DIM; ++j) {
            row[j] = hmm_logA[(l * S_DIM + s) * S_DIM + j];
            mx = fmaxf(mx, row[j]);
        }
        float sum = 0.f;
#pragma unroll
        for (int j = 0; j < S_DIM; ++j) sum += __expf(row[j] - mx);
        float lse = mx + __logf(sum);
#pragma unroll
        for (int j = 0; j < S_DIM; ++j) ldsT[l][s][j] = row[j] - lse;
    }
    __syncthreads();

    // log_softmax of initial logits within each 8-lane state group
    float lpi;
    {
        float raw = hmm_log_pi[l * S_DIM + s];
        float m = raw;
        m = fmaxf(m, __shfl_xor(m, 1, 8));
        m = fmaxf(m, __shfl_xor(m, 2, 8));
        m = fmaxf(m, __shfl_xor(m, 4, 8));
        float e = __expf(raw - m);
        float se = e;
        se += __shfl_xor(se, 1, 8);
        se += __shfl_xor(se, 2, 8);
        se += __shfl_xor(se, 4, 8);
        lpi = raw - (m + __logf(se));
    }

    // per-(l,s) diag-Gaussian emission params
    float mean[F_DIM], iv[F_DIM], cst = 0.f;
#pragma unroll
    for (int f = 0; f < F_DIM; ++f) {
        mean[f] = hmm_means[(l * S_DIM + s) * F_DIM + f];
        float lv = hmm_log_vars[(l * S_DIM + s) * F_DIM + f];
        iv[f] = __expf(-lv);
        cst += lv + LOG2PI;
    }

    // EMA state (lanes s<3 own feature f=s of layer l)
    const float a = ema_alphas[l];
    float em = 0.f, ev = 1.f;
    const bool isE = (s < F_DIM);

    float la = 0.f;
    for (int t = 0; t < T_DIM; ++t) {
        float xn = 0.f;
        if (isE) {
            float xe = x[(t * B_DIM + b) * F_DIM + s];
            em = (1.f - a) * em + a * xe;
            ev = (1.f - a) * ev + a * xe * xe;
            xn = (xe - em) * rsqrtf(fmaxf(ev - em * em, 1e-6f));
        }
        // broadcast standardized features of this layer to all 8 state lanes
        float x0 = __shfl(xn, 0, 8);
        float x1 = __shfl(xn, 1, 8);
        float x2 = __shfl(xn, 2, 8);
        float d0 = x0 - mean[0], d1 = x1 - mean[1], d2 = x2 - mean[2];
        float ll = -0.5f * (d0 * d0 * iv[0] + d1 * d1 * iv[1] + d2 * d2 * iv[2] + cst);

        float val;
        if (t == 0) {
            val = lpi + ll;
        } else {
            float terms[S_DIM];
            float mx = -1e30f;
#pragma unroll
            for (int i = 0; i < S_DIM; ++i) {
                float lai = __shfl(la, i, 8);
                terms[i] = lai + ldsT[l][i][s];
                mx = fmaxf(mx, terms[i]);
            }
            float sum = 0.f;
#pragma unroll
            for (int i = 0; i < S_DIM; ++i) sum += __expf(terms[i] - mx);
            val = mx + __logf(sum) + ll;
        }
        // log_softmax over the 8 states
        float gm = val;
        gm = fmaxf(gm, __shfl_xor(gm, 1, 8));
        gm = fmaxf(gm, __shfl_xor(gm, 2, 8));
        gm = fmaxf(gm, __shfl_xor(gm, 4, 8));
        float e = __expf(val - gm);
        float ss = e;
        ss += __shfl_xor(ss, 1, 8);
        ss += __shfl_xor(ss, 2, 8);
        ss += __shfl_xor(ss, 4, 8);
        la = val - gm - __logf(ss);

        float post = __expf(la);
        s_out[((t * B_DIM + b) * L_DIM + l) * S_DIM + s] = post;
        int hidx = (t * B_DIM + b) * D_DIM + s * L_DIM + l;   // 't b l s -> t b (s l)'
        h32[hidx] = post;
        h16[hidx] = (_Float16)post;
    }
}

// ---------------------------------------------------------------------------
// K2: xz = h @ W_in + b_in via WMMA 16x16x32 f16. Block = 8 waves (8 N-tiles)
// x 4 M-tiles; B fragment register-resident across the 4 wmmas. The x/z
// split is wave-uniform (waves 0-3 -> x, waves 4-7 -> silu(z)).
// ---------------------------------------------------------------------------
__global__ void gemm_in_kernel(const _Float16* __restrict__ h16,
                               const _Float16* __restrict__ win_frag,
                               const float* __restrict__ b_in,
                               float* __restrict__ x_in,
                               float* __restrict__ zbuf) {
    const int tid = threadIdx.x;
    const int wave = tid >> 5;
    const int lane = tid & 31;
    const int g = lane >> 4;
    const int n = lane & 15;
    const int gc = wave * 16 + n;

    v16h bm = load_b_frag(win_frag, wave, lane);
    const float bias = b_in[gc];

    if (wave < 4) {           // wave-uniform: x branch (cols 0..63)
#pragma unroll
        for (int m = 0; m < 4; ++m) {
            int rowbase = blockIdx.x * 64 + m * 16;
            v16h a = load_a_tile(h16, rowbase + n, D_DIM, 0, g);
            v8f c = {};
            c = __builtin_amdgcn_wmma_f32_16x16x32_f16(false, a, false, bm, (short)0, c, false, false);
#pragma unroll
            for (int v = 0; v < 8; ++v)
                x_in[(rowbase + v + 8 * g) * DI_DIM + gc] = c[v] + bias;
        }
    } else {                  // wave-uniform: z branch (cols 64..127)
        const int zc = gc - DI_DIM;
#pragma unroll
        for (int m = 0; m < 4; ++m) {
            int rowbase = blockIdx.x * 64 + m * 16;
            v16h a = load_a_tile(h16, rowbase + n, D_DIM, 0, g);
            v8f c = {};
            c = __builtin_amdgcn_wmma_f32_16x16x32_f16(false, a, false, bm, (short)0, c, false, false);
#pragma unroll
            for (int v = 0; v < 8; ++v)
                zbuf[(rowbase + v + 8 * g) * DI_DIM + zc] = siluf_(c[v] + bias);
        }
    }
}

// ---------------------------------------------------------------------------
// K3: depthwise causal conv (K=4) over time + silu
// ---------------------------------------------------------------------------
__global__ void conv_kernel(const float* __restrict__ x_in,
                            const float* __restrict__ conv_w,
                            const float* __restrict__ conv_b,
                            float* __restrict__ xss,
                            _Float16* __restrict__ xss16) {
    int idx = blockIdx.x * 256 + threadIdx.x;           // over TB*64
    int c = idx & (DI_DIM - 1);
    int tb = idx >> 6;
    int b = tb & (B_DIM - 1);
    int t = tb >> 3;
    float acc = conv_b[c];
#pragma unroll
    for (int k = 0; k < K_DIM; ++k) {
        int tt = t - (K_DIM - 1) + k;
        if (tt >= 0) acc = fmaf(x_in[(tt * B_DIM + b) * DI_DIM + c], conv_w[c * K_DIM + k], acc);
    }
    float v = siluf_(acc);
    xss[idx] = v;
    xss16[idx] = (_Float16)v;
}

// ---------------------------------------------------------------------------
// K4: xdb = xss @ W_x (N padded 34->48, K=64 -> 2 WMMAs). 3 waves = 3 N-tiles,
// 4 M-tiles per block; both B fragments register-resident.
// ---------------------------------------------------------------------------
__global__ void gemm_x_kernel(const _Float16* __restrict__ xss16,
                              const _Float16* __restrict__ wx_frag,
                              float* __restrict__ xdb) {
    const int tid = threadIdx.x;                         // 96 threads
    const int wave = tid >> 5;
    const int lane = tid & 31;
    const int g = lane >> 4;
    const int n = lane & 15;
    const int col = wave * 16 + n;

    v16h b0 = load_b_frag(wx_frag, wave * 2 + 0, lane);
    v16h b1 = load_b_frag(wx_frag, wave * 2 + 1, lane);

#pragma unroll
    for (int m = 0; m < 4; ++m) {
        int rowbase = blockIdx.x * 64 + m * 16;
        v8f c = {};
        v16h a0 = load_a_tile(xss16, rowbase + n, DI_DIM, 0, g);
        c = __builtin_amdgcn_wmma_f32_16x16x32_f16(false, a0, false, b0, (short)0, c, false, false);
        v16h a1 = load_a_tile(xss16, rowbase + n, DI_DIM, 32, g);
        c = __builtin_amdgcn_wmma_f32_16x16x32_f16(false, a1, false, b1, (short)0, c, false, false);
#pragma unroll
        for (int v = 0; v < 8; ++v)
            xdb[(rowbase + v + 8 * g) * 48 + col] = c[v];
    }
}

// ---------------------------------------------------------------------------
// K5: Mamba SSM recurrence. 8 blocks (one per batch) x 1024 threads:
// thread = (di, ds); ds is the low 4 lane bits so the state reduction is a
// 16-lane shfl_xor tree inside a half-wave. dt / dA computed on the fly,
// next-step operands software-pipelined, prefetch hint for t+8.
// ---------------------------------------------------------------------------
__global__ void ssm_scan_kernel(const float* __restrict__ xdb,
                                const float* __restrict__ xss,
                                const float* __restrict__ zbuf,
                                const float* __restrict__ W_dt,
                                const float* __restrict__ b_dt,
                                const float* __restrict__ A_log,
                                const float* __restrict__ D_skip,
                                _Float16* __restrict__ ym16) {
    const int b = blockIdx.x;
    const int tid = threadIdx.x;
    const int ds = tid & 15;
    const int di = tid >> 4;

    const float w0 = W_dt[di];
    const float w1 = W_dt[DI_DIM + di];
    const float bd = b_dt[di];
    const float Acoef = -__expf(A_log[di * DST_DIM + ds]);
    const float Dsk = D_skip[di];

    float hstate = 0.f;
    // prologue: load t=0 operands (zv loaded by all lanes -> uniform broadcast)
    int base = b;
    float d0 = xdb[base * 48 + 0], d1 = xdb[base * 48 + 1];
    float Bm = xdb[base * 48 + 2 + ds], Cm = xdb[base * 48 + 18 + ds];
    float xv = xss[base * 64 + di];
    float zv = zbuf[base * 64 + di];

    for (int t = 0; t < T_DIM; ++t) {
        float nd0 = 0.f, nd1 = 0.f, nBm = 0.f, nCm = 0.f, nxv = 0.f, nzv = 0.f;
        if (t + 1 < T_DIM) {
            int nb = (t + 1) * B_DIM + b;
            nd0 = xdb[nb * 48 + 0];
            nd1 = xdb[nb * 48 + 1];
            nBm = xdb[nb * 48 + 2 + ds];
            nCm = xdb[nb * 48 + 18 + ds];
            nxv = xss[nb * 64 + di];
            nzv = zbuf[nb * 64 + di];
        }
        if (t + 8 < T_DIM) {
            __builtin_prefetch(&xdb[((size_t)(t + 8) * B_DIM + b) * 48], 0, 1);
            __builtin_prefetch(&xss[((size_t)(t + 8) * B_DIM + b) * 64], 0, 1);
        }

        float dt = softplusf_(fmaf(d0, w0, fmaf(d1, w1, bd)));
        float dA = __expf(dt * Acoef);
        hstate = fmaf(dA, hstate, dt * Bm * xv);
        float contr = hstate * Cm;
        contr += __shfl_xor(contr, 1, 16);
        contr += __shfl_xor(contr, 2, 16);
        contr += __shfl_xor(contr, 4, 16);
        contr += __shfl_xor(contr, 8, 16);
        if (ds == 0) {
            int ob = t * B_DIM + b;
            float ymv = (contr + Dsk * xv) * zv;
            ym16[ob * DI_DIM + di] = (_Float16)ymv;
        }
        d0 = nd0; d1 = nd1; Bm = nBm; Cm = nCm; xv = nxv; zv = nzv;
    }
}

// ---------------------------------------------------------------------------
// K6: m_out = ym @ W_out + b_out (K=64 -> 2 WMMAs, 2 N-tiles -> 2 waves),
// 4 M-tiles per block.
// ---------------------------------------------------------------------------
__global__ void gemm_out_kernel(const _Float16* __restrict__ ym16,
                                const _Float16* __restrict__ wout_frag,
                                const float* __restrict__ b_out,
                                float* __restrict__ m_out) {
    const int tid = threadIdx.x;                         // 64 threads
    const int wave = tid >> 5;
    const int lane = tid & 31;
    const int g = lane >> 4;
    const int n = lane & 15;
    const int col = wave * 16 + n;

    v16h b0 = load_b_frag(wout_frag, wave * 2 + 0, lane);
    v16h b1 = load_b_frag(wout_frag, wave * 2 + 1, lane);
    const float bias = b_out[col];

#pragma unroll
    for (int m = 0; m < 4; ++m) {
        int rowbase = blockIdx.x * 64 + m * 16;
        v8f c = {};
        v16h a0 = load_a_tile(ym16, rowbase + n, DI_DIM, 0, g);
        c = __builtin_amdgcn_wmma_f32_16x16x32_f16(false, a0, false, b0, (short)0, c, false, false);
        v16h a1 = load_a_tile(ym16, rowbase + n, DI_DIM, 32, g);
        c = __builtin_amdgcn_wmma_f32_16x16x32_f16(false, a1, false, b1, (short)0, c, false, false);
#pragma unroll
        for (int v = 0; v < 8; ++v)
            m_out[(rowbase + v + 8 * g) * D_DIM + col] = c[v] + bias;
    }
}

// ---------------------------------------------------------------------------
// K7: residual + LayerNorm + sigmoid head (one thread per token)
// ---------------------------------------------------------------------------
__global__ void head_kernel(const float* __restrict__ h32,
                            const float* __restrict__ m_out,
                            const float* __restrict__ ln_scale,
                            const float* __restrict__ ln_bias,
                            const float* __restrict__ proj_w,
                            const float* __restrict__ proj_b,
                            float* __restrict__ y_out) {
    int idx = blockIdx.x * 256 + threadIdx.x;            // over TB
    float r[D_DIM];
    float mean = 0.f;
#pragma unroll
    for (int j = 0; j < D_DIM; ++j) {
        r[j] = h32[idx * D_DIM + j] + m_out[idx * D_DIM + j];
        mean += r[j];
    }
    mean *= (1.0f / D_DIM);
    float var = 0.f;
#pragma unroll
    for (int j = 0; j < D_DIM; ++j) {
        float d = r[j] - mean;
        var += d * d;
    }
    var *= (1.0f / D_DIM);
    float inv = rsqrtf(var + 1e-5f);
    float acc = proj_b[0];
#pragma unroll
    for (int j = 0; j < D_DIM; ++j) {
        float rn = (r[j] - mean) * inv * ln_scale[j] + ln_bias[j];
        acc = fmaf(rn, proj_w[j], acc);
    }
    y_out[idx] = sigmoidf_(acc);
}

// ---------------------------------------------------------------------------
// Launch
// ---------------------------------------------------------------------------
extern "C" void kernel_launch(void* const* d_in, const int* in_sizes, int n_in,
                              void* d_out, int out_size, void* d_ws, size_t ws_size,
                              hipStream_t stream) {
    const float* x            = (const float*)d_in[0];
    const float* ema_alphas   = (const float*)d_in[1];
    const float* hmm_logA     = (const float*)d_in[2];
    const float* hmm_log_pi   = (const float*)d_in[3];
    const float* hmm_means    = (const float*)d_in[4];
    const float* hmm_log_vars = (const float*)d_in[5];
    const float* W_in         = (const float*)d_in[6];
    const float* b_in         = (const float*)d_in[7];
    const float* conv_w       = (const float*)d_in[8];
    const float* conv_b       = (const float*)d_in[9];
    const float* W_x          = (const float*)d_in[10];
    const float* W_dt         = (const float*)d_in[11];
    const float* b_dt         = (const float*)d_in[12];
    const float* A_log        = (const float*)d_in[13];
    const float* D_skip       = (const float*)d_in[14];
    const float* W_out        = (const float*)d_in[15];
    const float* b_out        = (const float*)d_in[16];
    const float* ln_scale     = (const float*)d_in[17];
    const float* ln_bias      = (const float*)d_in[18];
    const float* proj_w       = (const float*)d_in[19];
    const float* proj_b       = (const float*)d_in[20];

    float* out   = (float*)d_out;
    float* s_out = out;                                        // [T,B,L,S]
    float* y_out = out + (size_t)TB_DIM * L_DIM * S_DIM;       // [T,B,1]

    char* ws = (char*)d_ws;
    float*    h32       = (float*)(ws + WS_H32);
    _Float16* h16       = (_Float16*)(ws + WS_H16);
    float*    x_in      = (float*)(ws + WS_XIN);
    float*    zbuf      = (float*)(ws + WS_ZB);
    float*    xss       = (float*)(ws + WS_XSS);
    _Float16* xss16     = (_Float16*)(ws + WS_XSS16);
    float*    xdb       = (float*)(ws + WS_XDB);
    _Float16* ym16      = (_Float16*)(ws + WS_YM16);
    float*    m_out     = (float*)(ws + WS_MOUT);
    _Float16* win_frag  = (_Float16*)(ws + WS_WIN16);
    _Float16* wx_frag   = (_Float16*)(ws + WS_WX16);
    _Float16* wout_frag = (_Float16*)(ws + WS_WOUT16);

    prep_weights_kernel<<<1, 256, 0, stream>>>(W_in, W_x, W_out, win_frag, wx_frag, wout_frag);

    ema_hmm_scan_kernel<<<B_DIM, 32, 0, stream>>>(x, ema_alphas, hmm_logA, hmm_log_pi,
                                                  hmm_means, hmm_log_vars,
                                                  s_out, h32, h16);

    gemm_in_kernel<<<TB_DIM / 64, 256, 0, stream>>>(h16, win_frag, b_in, x_in, zbuf);

    conv_kernel<<<(TB_DIM * DI_DIM) / 256, 256, 0, stream>>>(x_in, conv_w, conv_b, xss, xss16);

    gemm_x_kernel<<<TB_DIM / 64, 96, 0, stream>>>(xss16, wx_frag, xdb);

    ssm_scan_kernel<<<B_DIM, 1024, 0, stream>>>(xdb, xss, zbuf, W_dt, b_dt,
                                                A_log, D_skip, ym16);

    gemm_out_kernel<<<TB_DIM / 64, 64, 0, stream>>>(ym16, wout_frag, b_out, m_out);

    head_kernel<<<TB_DIM / 256, 256, 0, stream>>>(h32, m_out, ln_scale, ln_bias,
                                                  proj_w, proj_b, y_out);
}
